// MultiHeadedAttention_11493332484673
// MI455X (gfx1250) — compile-verified, matrix operand order checked by reading, not run on hardware
//
#include <hip/hip_runtime.h>
#include <hip/hip_bf16.h>

typedef __attribute__((ext_vector_type(16))) _Float16 v16h;
typedef __attribute__((ext_vector_type(8)))  _Float16 v8h;
typedef __attribute__((ext_vector_type(4)))  _Float16 v4h;
typedef __attribute__((ext_vector_type(8)))  float    v8f;

#define D_MODEL 512
#define N_HEADS 8
#define HEAD_DIM 64
#define BATCH 2
#define SEQ 4096
#define ROWS (BATCH * SEQ)   // 8192

// ---------------------------------------------------------------------------
// D = A(16x32 f16) * B(32x16 f16) + C(16x16 f32), wave32 CDNA5 WMMA
// ---------------------------------------------------------------------------
static __device__ __forceinline__ v8f wmma_f16(v16h a, v16h b, v8f c) {
  return __builtin_amdgcn_wmma_f32_16x16x32_f16(
      /*neg_a=*/false, a, /*neg_b=*/false, b,
      /*c_mod=*/(short)0, c, /*reuse_a=*/false, /*reuse_b=*/false);
}

// Load one lane's share of a 16-bit A/B fragment (16x32 / 32x16).
// ISA layout: lane holds K = k0..k0+7 in VGPRs 0-3 (pairs) and
// K = k0+16..k0+23 in VGPRs 4-7, where k0 = (lane>>4)*8.
// Two contiguous 16-byte loads per lane.
static __device__ __forceinline__ v16h load_frag(const _Float16* p, int k0) {
  v8h lo = *(const v8h*)(p + k0);
  v8h hi = *(const v8h*)(p + k0 + 16);
  v16h f;
#pragma unroll
  for (int i = 0; i < 8; ++i) { f[i] = lo[i]; f[i + 8] = hi[i]; }
  return f;
}

// ---------------------------------------------------------------------------
// fp32 -> f16 conversion (vectorized x4)
// ---------------------------------------------------------------------------
__global__ __launch_bounds__(256) void cvt_f32_f16(const float* __restrict__ src,
                                                   _Float16* __restrict__ dst,
                                                   int n4) {
  int i = blockIdx.x * blockDim.x + threadIdx.x;
  if (i < n4) {
    float4 v = ((const float4*)src)[i];
    v4h o;
    o[0] = (_Float16)v.x; o[1] = (_Float16)v.y;
    o[2] = (_Float16)v.z; o[3] = (_Float16)v.w;
    ((v4h*)dst)[i] = o;
  }
}

// ---------------------------------------------------------------------------
// Projection GEMM: Y = (X @ W^T + bias) * scale, X:[8192,512] f16, W:[512,512] f16
// One wave computes a 16x64 tile (4 N-subtiles). Software-pipelined K loop,
// unrolled x2 so the two fragment generations ping-pong between fixed
// register sets (no rotation movs) while loads overlap the WMMAs.
// grid = (64, 8), block = 256 (8 waves stacked along M -> 128x64 per block)
// vmode==0: store [B,H,S,Dh] (Q, K)     vmode==1: store transposed [B,H,Dh,S] (V)
// ---------------------------------------------------------------------------
__global__ __launch_bounds__(256) void proj_kernel(const _Float16* __restrict__ X,
                                                   const _Float16* __restrict__ W,
                                                   const float* __restrict__ bias,
                                                   _Float16* __restrict__ Y,
                                                   float scale, int vmode) {
  int wave = threadIdx.x >> 5, lane = threadIdx.x & 31;
  int mt = blockIdx.x * 8 + wave;   // 0..511  (M tiles of 16)
  int nb = blockIdx.y;              // 0..7    (N blocks of 64)
  int mn = lane & 15, half = lane >> 4, k0 = half * 8;

  const _Float16* xrow  = X + ((size_t)mt * 16 + mn) * D_MODEL;    // A row m
  const _Float16* wrow0 = W + ((size_t)(nb * 64 + 0 * 16) + mn) * D_MODEL;
  const _Float16* wrow1 = W + ((size_t)(nb * 64 + 1 * 16) + mn) * D_MODEL;
  const _Float16* wrow2 = W + ((size_t)(nb * 64 + 2 * 16) + mn) * D_MODEL;
  const _Float16* wrow3 = W + ((size_t)(nb * 64 + 3 * 16) + mn) * D_MODEL;

  v8f acc0, acc1, acc2, acc3;
  {
    float b0 = bias[nb * 64 + 0 * 16 + mn];
    float b1 = bias[nb * 64 + 1 * 16 + mn];
    float b2 = bias[nb * 64 + 2 * 16 + mn];
    float b3 = bias[nb * 64 + 3 * 16 + mn];
#pragma unroll
    for (int r = 0; r < 8; ++r) {
      acc0[r] = b0; acc1[r] = b1; acc2[r] = b2; acc3[r] = b3;
    }
  }

  // pipeline prologue: chunk 0 fragments
  v16h a  = load_frag(xrow,  k0);
  v16h b0 = load_frag(wrow0, k0);
  v16h b1 = load_frag(wrow1, k0);
  v16h b2 = load_frag(wrow2, k0);
  v16h b3 = load_frag(wrow3, k0);

#pragma unroll 2
  for (int kk = 0; kk < D_MODEL / 32; ++kk) {
    // issue next chunk's loads (wraps to 0 on last iter; results unused then)
    int kn = ((kk + 1) & (D_MODEL / 32 - 1)) * 32;
    v16h na  = load_frag(xrow  + kn, k0);
    v16h nb0 = load_frag(wrow0 + kn, k0);
    v16h nb1 = load_frag(wrow1 + kn, k0);
    v16h nb2 = load_frag(wrow2 + kn, k0);
    v16h nb3 = load_frag(wrow3 + kn, k0);
    // math on current fragments (loaded one iteration ago)
    acc0 = wmma_f16(a, b0, acc0);
    acc1 = wmma_f16(a, b1, acc1);
    acc2 = wmma_f16(a, b2, acc2);
    acc3 = wmma_f16(a, b3, acc3);
    a = na; b0 = nb0; b1 = nb1; b2 = nb2; b3 = nb3;
  }

  float accs[4][8];
#pragma unroll
  for (int r = 0; r < 8; ++r) {
    accs[0][r] = acc0[r]; accs[1][r] = acc1[r];
    accs[2][r] = acc2[r]; accs[3][r] = acc3[r];
  }

  int row0 = mt * 16 + half * 8;             // 8 consecutive output rows
#pragma unroll
  for (int t = 0; t < 4; ++t) {
    int col = nb * 64 + t * 16 + mn;
    int h = col >> 6, d = col & (HEAD_DIM - 1);
    if (vmode == 0) {
#pragma unroll
      for (int r = 0; r < 8; ++r) {
        int row = row0 + r;
        int b_ = row >> 12, s = row & (SEQ - 1);
        Y[(((size_t)(b_ * N_HEADS + h) * SEQ) + s) * HEAD_DIM + d] =
            (_Float16)(accs[t][r] * scale);
      }
    } else {
      // transposed: 8 rows are contiguous along S -> one 16-byte store
      int b_ = row0 >> 12, s0 = row0 & (SEQ - 1);
      v8h pk;
#pragma unroll
      for (int r = 0; r < 8; ++r) pk[r] = (_Float16)(accs[t][r] * scale);
      *(v8h*)(Y + (((size_t)(b_ * N_HEADS + h) * HEAD_DIM) + d) * SEQ + s0) = pk;
    }
  }
}

// ---------------------------------------------------------------------------
// Flash attention. Q,K: [B,H,S,64] f16 (Q pre-scaled by 1/8). Vt: [B,H,64,S].
// One wave handles a 16-query tile for one (b,h); 32 keys per iteration.
// K-tile fragments are software-pipelined across kt (unrolled x2 -> ping-pong
// registers, no rotation movs); V-tile loads issue before the softmax so the
// P.V WMMAs find them complete.
// grid = 512, block = 256 (8 waves)
// ---------------------------------------------------------------------------
__global__ __launch_bounds__(256) void flash_kernel(const _Float16* __restrict__ Q,
                                                    const _Float16* __restrict__ K,
                                                    const _Float16* __restrict__ Vt,
                                                    _Float16* __restrict__ ctx) {
  __shared__ _Float16 lds_p[8][16 * 32];   // per-wave P tile (row-major 16x32)
  int wave = threadIdx.x >> 5, lane = threadIdx.x & 31;
  int tile = blockIdx.x * 8 + wave;        // 0..4095
  int qt = tile & 255;                     // query tile within (b,h)
  int bh = tile >> 8;                      // 0..15
  const _Float16* Qbh = Q  + (size_t)bh * SEQ * HEAD_DIM;
  const _Float16* Kbh = K  + (size_t)bh * SEQ * HEAD_DIM;
  const _Float16* Vbh = Vt + (size_t)bh * HEAD_DIM * SEQ;
  int n = lane & 15, half = lane >> 4, k0 = half * 8;

  // Q tile resident in registers: 16x64 = two A fragments
  const _Float16* qrow = Qbh + ((size_t)qt * 16 + n) * HEAD_DIM;
  v16h qf0 = load_frag(qrow, k0);
  v16h qf1 = load_frag(qrow + 32, k0);

  v8f acc0 = {}, acc1 = {}, acc2 = {}, acc3 = {};
  float mrow[8], lrow[8];
#pragma unroll
  for (int r = 0; r < 8; ++r) { mrow[r] = -1e30f; lrow[r] = 0.0f; }

  _Float16* lp = lds_p[wave];

  // pipeline prologue: K fragments for kt = 0
  const _Float16* kr0 = Kbh + (size_t)n * HEAD_DIM;
  v16h kb00 = load_frag(kr0, k0);
  v16h kb01 = load_frag(kr0 + 32, k0);
  v16h kb10 = load_frag(kr0 + 16 * HEAD_DIM, k0);
  v16h kb11 = load_frag(kr0 + 16 * HEAD_DIM + 32, k0);

#pragma unroll 2
  for (int kt = 0; kt < SEQ / 32; ++kt) {
    // scores = Q (16x64) @ K^T (64x32) on current (pipelined) K fragments
    v8f s0 = {}, s1 = {};
    s0 = wmma_f16(qf0, kb00, s0);
    s0 = wmma_f16(qf1, kb01, s0);
    s1 = wmma_f16(qf0, kb10, s1);
    s1 = wmma_f16(qf1, kb11, s1);

    // issue NEXT K tile loads (wraps on last iteration; results unused then)
    {
      int ktn = (kt + 1) & (SEQ / 32 - 1);
      const _Float16* krn = Kbh + ((size_t)ktn * 32 + n) * HEAD_DIM;
      kb00 = load_frag(krn, k0);
      kb01 = load_frag(krn + 32, k0);
      kb10 = load_frag(krn + 16 * HEAD_DIM, k0);
      kb11 = load_frag(krn + 16 * HEAD_DIM + 32, k0);
    }

    // issue current V-tile loads early (overlap with the softmax below)
    const _Float16* vr = Vbh + (size_t)n * SEQ + (size_t)kt * 32;
    v16h vb0 = load_frag(vr, k0);                    vr += (size_t)16 * SEQ;
    v16h vb1 = load_frag(vr, k0);                    vr += (size_t)16 * SEQ;
    v16h vb2 = load_frag(vr, k0);                    vr += (size_t)16 * SEQ;
    v16h vb3 = load_frag(vr, k0);

    // online softmax: row stats via xor-shuffle across the 16-lane half-group
    float p0[8], p1[8], corr[8];
#pragma unroll
    for (int r = 0; r < 8; ++r) {
      float mx = fmaxf(s0[r], s1[r]);
      mx = fmaxf(mx, __shfl_xor(mx, 1));
      mx = fmaxf(mx, __shfl_xor(mx, 2));
      mx = fmaxf(mx, __shfl_xor(mx, 4));
      mx = fmaxf(mx, __shfl_xor(mx, 8));
      float nm = fmaxf(mrow[r], mx);
      corr[r] = __expf(mrow[r] - nm);
      mrow[r] = nm;
      p0[r] = __expf(s0[r] - nm);
      p1[r] = __expf(s1[r] - nm);
      float rs = p0[r] + p1[r];
      rs += __shfl_xor(rs, 1);
      rs += __shfl_xor(rs, 2);
      rs += __shfl_xor(rs, 4);
      rs += __shfl_xor(rs, 8);
      lrow[r] = lrow[r] * corr[r] + rs;
    }
#pragma unroll
    for (int r = 0; r < 8; ++r) {
      acc0[r] *= corr[r]; acc1[r] *= corr[r];
      acc2[r] *= corr[r]; acc3[r] *= corr[r];
    }

    // Transpose P (C layout, column-striped) -> A layout via LDS
#pragma unroll
    for (int r = 0; r < 8; ++r) {
      int mm = half * 8 + r;
      lp[mm * 32 + n]      = (_Float16)p0[r];
      lp[mm * 32 + 16 + n] = (_Float16)p1[r];
    }
    v16h pa = load_frag(lp + n * 32, k0);  // A-frag row m = lane&15

    // ctx += P (16x32) @ V (32x64)
    acc0 = wmma_f16(pa, vb0, acc0);
    acc1 = wmma_f16(pa, vb1, acc1);
    acc2 = wmma_f16(pa, vb2, acc2);
    acc3 = wmma_f16(pa, vb3, acc3);
  }

  // epilogue: normalize, store ctx merged-head [B,S,512] f16
  float inv_l[8];
#pragma unroll
  for (int r = 0; r < 8; ++r) inv_l[r] = 1.0f / lrow[r];
  int b_ = bh >> 3, h = bh & 7;
#pragma unroll
  for (int r = 0; r < 8; ++r) {
    int s = qt * 16 + half * 8 + r;
    size_t base = ((size_t)(b_ * SEQ + s)) * D_MODEL + h * HEAD_DIM + n;
    ctx[base +  0] = (_Float16)(acc0[r] * inv_l[r]);
    ctx[base + 16] = (_Float16)(acc1[r] * inv_l[r]);
    ctx[base + 32] = (_Float16)(acc2[r] * inv_l[r]);
    ctx[base + 48] = (_Float16)(acc3[r] * inv_l[r]);
  }
}

// ---------------------------------------------------------------------------
// Output projection: out = ctx @ Wo^T + bo, fp32 output.
// Same 16x64-per-wave register blocking + x2-unrolled software pipeline.
// grid = (64, 8)
// ---------------------------------------------------------------------------
__global__ __launch_bounds__(256) void oproj_kernel(const _Float16* __restrict__ X,
                                                    const _Float16* __restrict__ W,
                                                    const float* __restrict__ bias,
                                                    float* __restrict__ out) {
  int wave = threadIdx.x >> 5, lane = threadIdx.x & 31;
  int mt = blockIdx.x * 8 + wave;
  int nb = blockIdx.y;
  int mn = lane & 15, half = lane >> 4, k0 = half * 8;

  const _Float16* xrow  = X + ((size_t)mt * 16 + mn) * D_MODEL;
  const _Float16* wrow0 = W + ((size_t)(nb * 64 + 0 * 16) + mn) * D_MODEL;
  const _Float16* wrow1 = W + ((size_t)(nb * 64 + 1 * 16) + mn) * D_MODEL;
  const _Float16* wrow2 = W + ((size_t)(nb * 64 + 2 * 16) + mn) * D_MODEL;
  const _Float16* wrow3 = W + ((size_t)(nb * 64 + 3 * 16) + mn) * D_MODEL;

  v8f acc0, acc1, acc2, acc3;
  {
    float b0 = bias[nb * 64 + 0 * 16 + mn];
    float b1 = bias[nb * 64 + 1 * 16 + mn];
    float b2 = bias[nb * 64 + 2 * 16 + mn];
    float b3 = bias[nb * 64 + 3 * 16 + mn];
#pragma unroll
    for (int r = 0; r < 8; ++r) {
      acc0[r] = b0; acc1[r] = b1; acc2[r] = b2; acc3[r] = b3;
    }
  }

  v16h a  = load_frag(xrow,  k0);
  v16h b0 = load_frag(wrow0, k0);
  v16h b1 = load_frag(wrow1, k0);
  v16h b2 = load_frag(wrow2, k0);
  v16h b3 = load_frag(wrow3, k0);

#pragma unroll 2
  for (int kk = 0; kk < D_MODEL / 32; ++kk) {
    int kn = ((kk + 1) & (D_MODEL / 32 - 1)) * 32;
    v16h na  = load_frag(xrow  + kn, k0);
    v16h nb0 = load_frag(wrow0 + kn, k0);
    v16h nb1 = load_frag(wrow1 + kn, k0);
    v16h nb2 = load_frag(wrow2 + kn, k0);
    v16h nb3 = load_frag(wrow3 + kn, k0);
    acc0 = wmma_f16(a, b0, acc0);
    acc1 = wmma_f16(a, b1, acc1);
    acc2 = wmma_f16(a, b2, acc2);
    acc3 = wmma_f16(a, b3, acc3);
    a = na; b0 = nb0; b1 = nb1; b2 = nb2; b3 = nb3;
  }

  int row0 = mt * 16 + half * 8;
#pragma unroll
  for (int r = 0; r < 8; ++r) {
    size_t rb = (size_t)(row0 + r) * D_MODEL + nb * 64 + mn;
    out[rb +  0] = acc0[r];
    out[rb + 16] = acc1[r];
    out[rb + 32] = acc2[r];
    out[rb + 48] = acc3[r];
  }
}

// ---------------------------------------------------------------------------
extern "C" void kernel_launch(void* const* d_in, const int* in_sizes, int n_in,
                              void* d_out, int out_size, void* d_ws, size_t ws_size,
                              hipStream_t stream) {
  (void)in_sizes; (void)n_in; (void)out_size; (void)ws_size;
  const float* x  = (const float*)d_in[0];
  const float* Wq = (const float*)d_in[1];
  const float* bq = (const float*)d_in[2];
  const float* Wk = (const float*)d_in[3];
  const float* bk = (const float*)d_in[4];
  const float* Wv = (const float*)d_in[5];
  const float* bv = (const float*)d_in[6];
  const float* Wo = (const float*)d_in[7];
  const float* bo = (const float*)d_in[8];
  float* out = (float*)d_out;

  const size_t XS = (size_t)ROWS * D_MODEL;      // 4Mi halves per activation
  const size_t WS = (size_t)D_MODEL * D_MODEL;   // 256Ki halves per weight
  _Float16* x16  = (_Float16*)d_ws;
  _Float16* Wq16 = x16 + XS;
  _Float16* Wk16 = Wq16 + WS;
  _Float16* Wv16 = Wk16 + WS;
  _Float16* Wo16 = Wv16 + WS;
  _Float16* Q16  = Wo16 + WS;   // [B,H,S,Dh], pre-scaled by 1/8
  _Float16* K16  = Q16 + XS;    // [B,H,S,Dh]
  _Float16* Vt16 = K16 + XS;    // [B,H,Dh,S]
  _Float16* C16  = Vt16 + XS;   // [B,S,512]

  // fp32 -> f16 converts
  {
    int n4 = (int)(XS / 4);
    cvt_f32_f16<<<(n4 + 255) / 256, 256, 0, stream>>>(x, x16, n4);
    int w4 = (int)(WS / 4);
    cvt_f32_f16<<<(w4 + 255) / 256, 256, 0, stream>>>(Wq, Wq16, w4);
    cvt_f32_f16<<<(w4 + 255) / 256, 256, 0, stream>>>(Wk, Wk16, w4);
    cvt_f32_f16<<<(w4 + 255) / 256, 256, 0, stream>>>(Wv, Wv16, w4);
    cvt_f32_f16<<<(w4 + 255) / 256, 256, 0, stream>>>(Wo, Wo16, w4);
  }

  dim3 g(ROWS / 128, D_MODEL / 64), blk(256);
  const float qscale = 0.125f;  // 1/sqrt(HEAD_DIM)
  proj_kernel<<<g, blk, 0, stream>>>(x16, Wq16, bq, Q16, qscale, 0);
  proj_kernel<<<g, blk, 0, stream>>>(x16, Wk16, bk, K16, 1.0f, 0);
  proj_kernel<<<g, blk, 0, stream>>>(x16, Wv16, bv, Vt16, 1.0f, 1);

  flash_kernel<<<(BATCH * N_HEADS * (SEQ / 16)) / 8, 256, 0, stream>>>(
      Q16, K16, Vt16, C16);

  oproj_kernel<<<g, blk, 0, stream>>>(C16, Wo16, bo, out);
}
